// RayObsGraph_46815143526666
// MI455X (gfx1250) — matrix operand order repaired and checked by reading, not compiled, not run on hardware
//
#include <hip/hip_runtime.h>
#include <hip/hip_bf16.h>

// Problem constants (match reference)
#define BB    8
#define NNODE 2048
#define EEDGE 16384
#define OBSF  128
#define HF    256
#define OUTF  1024

typedef __bf16 bf16t;
typedef __attribute__((ext_vector_type(16))) __bf16 v16bf;
typedef __attribute__((ext_vector_type(8)))  __bf16 v8bf;
typedef __attribute__((ext_vector_type(8)))  float  v8f;

// fp32 -> bf16 round-to-nearest-even
__device__ __forceinline__ bf16t f2bf(float f) {
  union { float f; unsigned u; } c; c.f = f;
  unsigned r = c.u + 0x7FFFu + ((c.u >> 16) & 1u);
  unsigned short s = (unsigned short)(r >> 16);
  return __builtin_bit_cast(bf16t, s);
}

__device__ __forceinline__ v16bf load_frag(const bf16t* p) {
  v8bf lo = *(const v8bf*)p;
  v8bf hi = *(const v8bf*)(p + 8);
  return __builtin_shufflevector(lo, hi, 0, 1, 2, 3, 4, 5, 6, 7,
                                 8, 9, 10, 11, 12, 13, 14, 15);
}

// ---------------------------------------------------------------------------
// CSR build: histogram of dst, per-graph exclusive scan, fill src lists.
__global__ void k_zero_i32(int* __restrict__ p, int n) {
  int i = blockIdx.x * blockDim.x + threadIdx.x;
  if (i < n) p[i] = 0;
}

__global__ void k_count_edges(const int* __restrict__ edges, int* __restrict__ counts) {
  int i = blockIdx.x * blockDim.x + threadIdx.x;   // b*E + e
  if (i >= BB * EEDGE) return;
  int b = i / EEDGE, e = i - b * EEDGE;
  int dst = edges[(b * 2 + 1) * EEDGE + e];
  atomicAdd(&counts[b * NNODE + dst], 1);          // native u32 atomic
}

__global__ void k_dinv_from_counts(const int* __restrict__ counts, float* __restrict__ dinv) {
  int i = blockIdx.x * blockDim.x + threadIdx.x;
  if (i < BB * NNODE) dinv[i] = rsqrtf((float)counts[i] + 1.0f);  // +1 self loop
}

// One block (256 threads) per graph: exclusive scan of 2048 counts.
__global__ __launch_bounds__(256) void k_scan(const int* __restrict__ counts,
                                              int* __restrict__ offsets) {
  __shared__ int part[256];
  int b = blockIdx.x;
  int t = threadIdx.x;
  const int* c = counts + b * NNODE;
  int loc[8];
  int s = 0;
#pragma unroll
  for (int j = 0; j < 8; ++j) { loc[j] = c[t * 8 + j]; s += loc[j]; }
  part[t] = s;
  __syncthreads();
  for (int off = 1; off < 256; off <<= 1) {   // Hillis-Steele inclusive scan
    int add = (t >= off) ? part[t - off] : 0;
    __syncthreads();
    part[t] += add;
    __syncthreads();
  }
  int excl = part[t] - s;                     // exclusive prefix of this chunk
  int base = b * EEDGE + excl;                // graph b owns elist[b*E .. (b+1)*E)
  int run = 0;
#pragma unroll
  for (int j = 0; j < 8; ++j) {
    offsets[b * NNODE + t * 8 + j] = base + run;
    run += loc[j];
  }
  if (b == BB - 1 && t == 255) offsets[BB * NNODE] = BB * EEDGE;
}

__global__ void k_copy_i32(const int* __restrict__ src, int* __restrict__ dst, int n) {
  int i = blockIdx.x * blockDim.x + threadIdx.x;
  if (i < n) dst[i] = src[i];
}

__global__ void k_fill_elist(const int* __restrict__ edges, int* __restrict__ cursor,
                             int* __restrict__ elist) {
  int i = blockIdx.x * blockDim.x + threadIdx.x;   // b*E + e
  if (i >= BB * EEDGE) return;
  int b = i / EEDGE, e = i - b * EEDGE;
  int src = edges[(b * 2 + 0) * EEDGE + e];
  int dst = edges[(b * 2 + 1) * EEDGE + e];
  int pos = atomicAdd(&cursor[b * NNODE + dst], 1);
  elist[pos] = b * NNODE + src;                    // store global src node id
}

// ---------------------------------------------------------------------------
// Gather aggregation (no atomics): z[d] = dinv[d]*(dinv[d]*x[d] + sum dinv[s]*x[s])
__global__ void k_aggregate(const float* __restrict__ x, const float* __restrict__ dinv,
                            const int* __restrict__ offsets, const int* __restrict__ elist,
                            float* __restrict__ z, int F) {
  int d = blockIdx.x;                // global node id (uniform per block)
  int f = threadIdx.x;               // F threads per block
  int beg = offsets[d], end = offsets[d + 1];
  float dv = dinv[d];
  float acc = dv * x[(size_t)d * F + f];
  for (int e = beg; e < end; ++e) {  // uniform loop -> scalarized index loads
    int s = elist[e];
    acc += dinv[s] * x[(size_t)s * F + f];
  }
  z[(size_t)d * F + f] = dv * acc;
}

__global__ void k_f32_to_bf16(const float* __restrict__ in, bf16t* __restrict__ out, int n) {
  int i = blockIdx.x * blockDim.x + threadIdx.x;
  if (i < n) out[i] = f2bf(in[i]);
}

// ---------------------------------------------------------------------------
// Pack W[K x NOUT] (row-major fp32) into per-lane WMMA B-fragment layout (bf16).
__global__ void k_pack_w(const float* __restrict__ W, bf16t* __restrict__ Wp,
                         int K, int NOUT) {
  int idx = blockIdx.x * blockDim.x + threadIdx.x;
  if (idx >= K * NOUT) return;
  int j    = idx & 15;
  int lane = (idx >> 4) & 31;
  int rest = idx >> 9;
  int nkt  = K >> 5;
  int kt   = rest % nkt;
  int nt   = rest / nkt;
  int n = nt * 16 + (lane & 15);
  int h = lane >> 4;
  int k = kt * 32 + ((j < 8) ? (8 * h + j) : (16 + 8 * h + (j - 8)));
  Wp[idx] = f2bf(W[k * NOUT + n]);
}

// ---------------------------------------------------------------------------
// GEMM + bias + ReLU on WMMA. Each wave computes a 32(M) x 64(N) tile:
// 8 f32 accumulators; each B fragment feeds TWO WMMAs (two independent A
// streams), halving B load traffic per matrix op and overlapping waits.
template <int K, int NOUT>
__global__ __launch_bounds__(128) void k_gemm_bias_relu(
    const bf16t* __restrict__ A, const bf16t* __restrict__ Wp,
    const float* __restrict__ bias, float* __restrict__ out) {
  constexpr int NKT  = K / 32;
  constexpr int NGRP = NOUT / 64;
  int lane = threadIdx.x & 31;
  int wave = threadIdx.x >> 5;
  int work = blockIdx.x * 4 + wave;
  int mgrp = work / NGRP;          // 32-row group
  int ngrp = work - mgrp * NGRP;   // 64-col group
  int h  = lane >> 4;
  int lm = lane & 15;

  const bf16t* arow0 = A + (size_t)(mgrp * 32 + lm) * K + 8 * h;
  const bf16t* arow1 = arow0 + (size_t)16 * K;
  const bf16t* bbase = Wp + (size_t)((ngrp * 4) * NKT * 32 + lane) * 16;

  v8f acc[8] = {};
#pragma unroll
  for (int kt = 0; kt < NKT; ++kt) {
    v16bf a0 = __builtin_shufflevector(*(const v8bf*)(arow0 + kt * 32), *(const v8bf*)(arow0 + kt * 32 + 16), 0, 1, 2, 3, 4, 5, 6, 7, 8, 9, 10, 11, 12, 13, 14, 15);
    v16bf a1 = __builtin_shufflevector(*(const v8bf*)(arow1 + kt * 32), *(const v8bf*)(arow1 + kt * 32 + 16), 0, 1, 2, 3, 4, 5, 6, 7, 8, 9, 10, 11, 12, 13, 14, 15);
#pragma unroll
    for (int i = 0; i < 4; ++i) {
      v16bf b = load_frag(bbase + (size_t)(i * NKT + kt) * 32 * 16);
      acc[i]     = __builtin_amdgcn_wmma_f32_16x16x32_bf16(
          false, a0, false, b, (short)0, acc[i], false, false);
      acc[4 + i] = __builtin_amdgcn_wmma_f32_16x16x32_bf16(
          false, a1, false, b, (short)0, acc[4 + i], false, false);
    }
  }

#pragma unroll
  for (int m = 0; m < 2; ++m) {
    int rbase = mgrp * 32 + m * 16 + h * 8;  // D layout: lane half -> rows 0-7/8-15
#pragma unroll
    for (int i = 0; i < 4; ++i) {
      int col = (ngrp * 4 + i) * 16 + lm;
      float bb = bias[col];
#pragma unroll
      for (int v = 0; v < 8; ++v) {
        float val = acc[m * 4 + i][v] + bb;
        out[(size_t)(rbase + v) * NOUT + col] = val > 0.0f ? val : 0.0f;
      }
    }
  }
}

// ---------------------------------------------------------------------------
extern "C" void kernel_launch(void* const* d_in, const int* in_sizes, int n_in,
                              void* d_out, int out_size, void* d_ws, size_t ws_size,
                              hipStream_t stream) {
  (void)in_sizes; (void)n_in; (void)out_size; (void)ws_size;
  const float* nodes = (const float*)d_in[0];   // [B,N,128] f32
  const int*   edges = (const int*)d_in[1];     // [B,2,E]   i32
  const float* W0    = (const float*)d_in[2];   // [128,256]
  const float* b0    = (const float*)d_in[3];   // [256]
  const float* W1    = (const float*)d_in[4];   // [256,1024]
  const float* b1    = (const float*)d_in[5];   // [1024]
  float*       outp  = (float*)d_out;           // [B,N,1024] f32

  char* ws = (char*)d_ws;
  size_t off = 0;
  int*   counts  = (int*)(ws + off);   off += (size_t)BB * NNODE * 4;        // 64 KB
  int*   offsets = (int*)(ws + off);   off += ((size_t)BB * NNODE + 16) * 4; // 64 KB
  int*   cursor  = (int*)(ws + off);   off += (size_t)BB * NNODE * 4;        // 64 KB
  int*   elist   = (int*)(ws + off);   off += (size_t)BB * EEDGE * 4;        // 512 KB
  float* dinv    = (float*)(ws + off); off += (size_t)BB * NNODE * 4;        // 64 KB
  float* z0      = (float*)(ws + off); off += (size_t)BB * NNODE * OBSF * 4; // 8 MB
  float* x1      = (float*)(ws + off); off += (size_t)BB * NNODE * HF * 4;   // 16 MB
  float* z1      = (float*)(ws + off); off += (size_t)BB * NNODE * HF * 4;   // 16 MB
  bf16t* z0b     = (bf16t*)(ws + off); off += (size_t)BB * NNODE * OBSF * 2; // 4 MB
  bf16t* z1b     = (bf16t*)(ws + off); off += (size_t)BB * NNODE * HF * 2;   // 8 MB
  bf16t* W0p     = (bf16t*)(ws + off); off += (size_t)OBSF * HF * 2;         // 64 KB
  bf16t* W1p     = (bf16t*)(ws + off); off += (size_t)HF * OUTF * 2;         // 512 KB

  const int NE = BB * EEDGE, NV = BB * NNODE;

  // --- CSR build + dinv ---
  k_zero_i32<<<(NV + 255) / 256, 256, 0, stream>>>(counts, NV);
  k_count_edges<<<(NE + 255) / 256, 256, 0, stream>>>(edges, counts);
  k_dinv_from_counts<<<(NV + 255) / 256, 256, 0, stream>>>(counts, dinv);
  k_scan<<<BB, 256, 0, stream>>>(counts, offsets);
  k_copy_i32<<<(NV + 255) / 256, 256, 0, stream>>>(offsets, cursor, NV);
  k_fill_elist<<<(NE + 255) / 256, 256, 0, stream>>>(edges, cursor, elist);

  // --- pack weights into WMMA B-fragment layout ---
  k_pack_w<<<(OBSF * HF + 255) / 256, 256, 0, stream>>>(W0, W0p, OBSF, HF);
  k_pack_w<<<(HF * OUTF + 255) / 256, 256, 0, stream>>>(W1, W1p, HF, OUTF);

  // --- layer 0: gather-aggregate(128) -> GEMM(128->256) + bias + relu ---
  k_aggregate<<<NV, OBSF, 0, stream>>>(nodes, dinv, offsets, elist, z0, OBSF);
  k_f32_to_bf16<<<(NV * OBSF + 255) / 256, 256, 0, stream>>>(z0, z0b, NV * OBSF);
  k_gemm_bias_relu<OBSF, HF><<<(NV / 32) * (HF / 64) / 4, 128, 0, stream>>>(z0b, W0p, b0, x1);

  // --- layer 1: gather-aggregate(256) -> GEMM(256->1024) + bias + relu ---
  k_aggregate<<<NV, HF, 0, stream>>>(x1, dinv, offsets, elist, z1, HF);
  k_f32_to_bf16<<<(NV * HF + 255) / 256, 256, 0, stream>>>(z1, z1b, NV * HF);
  k_gemm_bias_relu<HF, OUTF><<<(NV / 32) * (OUTF / 64) / 4, 128, 0, stream>>>(z1b, W1p, b1, outp);
}
